// M_GFAM_52029233824340
// MI455X (gfx1250) — matrix operand order repaired and checked by reading, not compiled, not run on hardware
//
#include <hip/hip_runtime.h>

#define BB 4
#define CC 64
#define HH 160
#define WWD 160
#define NP (HH * WWD)        // 25600 pixels
#define BNROWS (BB * NP)     // 102400 gemm rows
#define ROWTILES (BNROWS / 16)       // 6400 waves: one 16-row x 64-col strip per wave
#define GEMM_BLOCKS (ROWTILES / 4)   // 1600 blocks of 128 threads (4 waves)

typedef __attribute__((ext_vector_type(16))) __bf16 v16bf;
typedef __attribute__((ext_vector_type(8)))  __bf16 v8bf;
typedef __attribute__((ext_vector_type(8)))  float  v8f;

static inline int cdiv(int a, int b) { return (a + b - 1) / b; }

// ===================== setup / elementwise kernels =====================

__global__ void k_deg_init(float* __restrict__ deg) {
    int i = blockIdx.x * blockDim.x + threadIdx.x;
    if (i < NP) deg[i] = 1.0f;
}

__global__ void k_deg_count(float* __restrict__ deg, const int* __restrict__ dst, int E) {
    int i = blockIdx.x * blockDim.x + threadIdx.x;
    if (i < E) atomicAdd(&deg[dst[i]], 1.0f);
}

// deg -> dis (in place), snorm = dis*dis
__global__ void k_norm_fin(float* __restrict__ deg, float* __restrict__ snorm) {
    int i = blockIdx.x * blockDim.x + threadIdx.x;
    if (i < NP) {
        float r = rsqrtf(deg[i]);
        deg[i] = r;
        snorm[i] = r * r;
    }
}

__global__ void k_zero_cnt(int* __restrict__ cnt) {
    int i = blockIdx.x * blockDim.x + threadIdx.x;
    if (i < NP) cnt[i] = 0;
}

// Bounded-degree (<=8) incoming adjacency: nidx[d][k]=src, nw[d][k]=dis[src]*dis[d]
__global__ void k_adj_build(const int* __restrict__ src, const int* __restrict__ dst,
                            const float* __restrict__ dis,
                            int* __restrict__ cnt, int* __restrict__ nidx,
                            float* __restrict__ nw, int E) {
    int e = blockIdx.x * blockDim.x + threadIdx.x;
    if (e >= E) return;
    int s = src[e], d = dst[e];
    int pos = atomicAdd(&cnt[d], 1);
    if (pos < 8) {
        nidx[d * 8 + pos] = s;
        nw[d * 8 + pos] = dis[s] * dis[d];
    }
}

// gcn_w: [27][in=64][out=64] fp32  ->  wt: [27][out][in] bf16
__global__ void k_cvt_gcn_w(const float* __restrict__ w, __bf16* __restrict__ wt) {
    int t = blockIdx.x * blockDim.x + threadIdx.x;
    if (t >= 27 * 64 * 64) return;
    int o  = t & 63;
    int ii = (t >> 6) & 63;
    int m  = t >> 12;
    wt[(m * 64 + o) * 64 + ii] = (__bf16)w[t];   // w[t] = w[m][ii][o]
}

// cw: [3][out=64][in=192] fp32 -> bf16 (already [out][in])
__global__ void k_cvt_cw(const float* __restrict__ w, __bf16* __restrict__ wt) {
    int t = blockIdx.x * blockDim.x + threadIdx.x;
    if (t >= 3 * 64 * 192) return;
    wt[t] = (__bf16)w[t];
}

// rb_w: [16][o=64][i=64][3][3] fp32 -> wt: [conv][tap=9][o][i] bf16
__global__ void k_cvt_rb(const float* __restrict__ w, __bf16* __restrict__ wt) {
    int t = blockIdx.x * blockDim.x + threadIdx.x;
    if (t >= 16 * 9 * 64 * 64) return;
    int ii   = t & 63;
    int o    = (t >> 6) & 63;
    int tap  = (t >> 12) % 9;
    int conv = t / (9 * 64 * 64);
    wt[t] = (__bf16)w[((conv * 64 + o) * 64 + ii) * 9 + tap];
}

// x [B][C][H][W] -> A [B*N][C] f32 + Ah bf16 mirror
__global__ void k_nchw2rows(const float* __restrict__ x, float* __restrict__ A,
                            __bf16* __restrict__ Ah) {
    int t = blockIdx.x * blockDim.x + threadIdx.x;
    if (t >= BB * CC * NP) return;
    int pp = t % NP;
    int bc = t / NP;
    int c = bc % CC, b = bc / CC;
    float v = x[t];
    size_t o = ((size_t)(b * NP + pp)) * CC + c;
    A[o] = v;
    Ah[o] = (__bf16)v;
}

__global__ void k_rows2nchw(const float* __restrict__ A, float* __restrict__ y) {
    int t = blockIdx.x * blockDim.x + threadIdx.x;
    if (t >= BB * CC * NP) return;
    int pp = t % NP;
    int bc = t / NP;
    int c = bc % CC, b = bc / CC;
    y[t] = A[((size_t)(b * NP + pp)) * CC + c];
}

// Fused GCN aggregation (all math f32): out_bf16 = relu(hw*snorm + bias + sum hw[src]*w)
__global__ void k_gcn_agg(const float* __restrict__ hw, const float* __restrict__ snorm,
                          const float* __restrict__ bias,
                          const int* __restrict__ nidx, const float* __restrict__ nw,
                          const int* __restrict__ cnt,
                          __bf16* __restrict__ out, int ldo, int colOff) {
    int t = blockIdx.x * blockDim.x + threadIdx.x;
    if (t >= BNROWS * CC) return;
    int c = t & 63;
    int rn = t >> 6;
    int n = rn % NP;
    int b = rn / NP;
    float val = hw[t] * snorm[n] + bias[c];
    int k = cnt[n];
    if (k > 8) k = 8;
    for (int i = 0; i < k; ++i) {
        int s = nidx[n * 8 + i];
        val += hw[((size_t)b * NP + s) * CC + c] * nw[n * 8 + i];
    }
    out[(size_t)rn * ldo + colOff + c] = (__bf16)(val > 0.f ? val : 0.f);
}

// ============ WMMA GEMM, 16 rows x 64 cols per wave (4 accumulators) ============
// Xh: [rows x K] bf16 (leading dim ldx); Wt: [64 x K] bf16 row-major.
// Outputs: Y f32 [rows x 64] (nullable), Yh bf16 mirror (nullable).
__global__ __launch_bounds__(128)
void k_gemm2(const __bf16* __restrict__ Xh, int ldx,
             const __bf16* __restrict__ Wt,
             float* __restrict__ Y,
             __bf16* __restrict__ Yh,
             const float* __restrict__ bias,
             const float* __restrict__ res,
             int K) {
    int wave = (blockIdx.x * blockDim.x + threadIdx.x) >> 5;
    int lane = threadIdx.x & 31;
    if (wave >= ROWTILES) return;

    int half = lane >> 4;        // A: K 0-15 vs 16-31 ; B: K 0-15 vs 16-31
    int m = lane & 15;
    int row = wave * 16 + m;

    const __bf16* arow = Xh + (size_t)row * ldx + half * 16;

    v8f acc0 = {}, acc1 = {}, acc2 = {}, acc3 = {};
    for (int k = 0; k < K; k += 32) {
        v16bf a;
        v8bf alo = *(const v8bf*)(arow + k);
        v8bf ahi = *(const v8bf*)(arow + k + 8);
#pragma unroll
        for (int i = 0; i < 8; ++i) { a[i] = alo[i]; a[8 + i] = ahi[i]; }
        const __bf16* wbase = Wt + (size_t)m * K + half * 16 + k;
        v16bf b0 = *(const v16bf*)(wbase);
        v16bf b1 = *(const v16bf*)(wbase + (size_t)16 * K);
        v16bf b2 = *(const v16bf*)(wbase + (size_t)32 * K);
        v16bf b3 = *(const v16bf*)(wbase + (size_t)48 * K);
        acc0 = __builtin_amdgcn_wmma_f32_16x16x32_bf16(false, a, false, b0, (short)0, acc0, false, false);
        acc1 = __builtin_amdgcn_wmma_f32_16x16x32_bf16(false, a, false, b1, (short)0, acc1, false, false);
        acc2 = __builtin_amdgcn_wmma_f32_16x16x32_bf16(false, a, false, b2, (short)0, acc2, false, false);
        acc3 = __builtin_amdgcn_wmma_f32_16x16x32_bf16(false, a, false, b3, (short)0, acc3, false, false);
    }

    int rowBase = wave * 16 + half * 8;
    v8f accs[4] = {acc0, acc1, acc2, acc3};
#pragma unroll
    for (int t = 0; t < 4; ++t) {
        int col = t * 16 + m;
        float bv = bias ? bias[col] : 0.f;
#pragma unroll
        for (int v = 0; v < 8; ++v) {
            int r = rowBase + v;
            float val = accs[t][v] + bv;
            if (res) val += res[(size_t)r * CC + col];
            if (Y)  Y[(size_t)r * CC + col] = val;
            if (Yh) Yh[(size_t)r * CC + col] = (__bf16)val;
        }
    }
}

// ============ WMMA conv3x3, 16 pixels x 64 outchannels per wave ============
__global__ __launch_bounds__(128)
void k_conv2(const __bf16* __restrict__ Xh,    // [BN x 64] bf16
             const __bf16* __restrict__ Wt,    // [9][64 out][64 in] bf16
             float* __restrict__ Y,            // f32 out (nullable)
             __bf16* __restrict__ Yh,          // bf16 mirror out (nullable)
             const float* __restrict__ bias,   // [64]
             const float* __restrict__ alpha,  // PReLU slope ptr or null
             const float* __restrict__ res) {  // f32 residual or null
    int wave = (blockIdx.x * blockDim.x + threadIdx.x) >> 5;
    int lane = threadIdx.x & 31;
    if (wave >= ROWTILES) return;

    int half = lane >> 4;
    int m = lane & 15;
    int row = wave * 16 + m;
    int b = row / NP;
    int pp = row - b * NP;
    int py = pp / WWD, px = pp - py * WWD;

    v8f acc0 = {}, acc1 = {}, acc2 = {}, acc3 = {};
    for (int t9 = 0; t9 < 9; ++t9) {
        int dy = t9 / 3 - 1, dx = t9 % 3 - 1;
        int yy = py + dy, xx = px + dx;
        bool valid = (yy >= 0) && (yy < HH) && (xx >= 0) && (xx < WWD);
        const __bf16* arow = nullptr;
        if (valid)
            arow = Xh + ((size_t)b * NP + (size_t)yy * WWD + (size_t)xx) * CC + half * 16;
#pragma unroll
        for (int k = 0; k < 64; k += 32) {
            v16bf a = {};
            if (valid) {
                v8bf alo = *(const v8bf*)(arow + k);
                v8bf ahi = *(const v8bf*)(arow + k + 8);
#pragma unroll
                for (int i = 0; i < 8; ++i) { a[i] = alo[i]; a[8 + i] = ahi[i]; }
            }
            const __bf16* wbase = Wt + ((size_t)t9 * 64 + m) * 64 + half * 16 + k;
            v16bf b0 = *(const v16bf*)(wbase);
            v16bf b1 = *(const v16bf*)(wbase + 16 * 64);
            v16bf b2 = *(const v16bf*)(wbase + 32 * 64);
            v16bf b3 = *(const v16bf*)(wbase + 48 * 64);
            acc0 = __builtin_amdgcn_wmma_f32_16x16x32_bf16(false, a, false, b0, (short)0, acc0, false, false);
            acc1 = __builtin_amdgcn_wmma_f32_16x16x32_bf16(false, a, false, b1, (short)0, acc1, false, false);
            acc2 = __builtin_amdgcn_wmma_f32_16x16x32_bf16(false, a, false, b2, (short)0, acc2, false, false);
            acc3 = __builtin_amdgcn_wmma_f32_16x16x32_bf16(false, a, false, b3, (short)0, acc3, false, false);
        }
    }

    float al = alpha ? alpha[0] : 0.f;
    int rowBase = wave * 16 + half * 8;
    v8f accs[4] = {acc0, acc1, acc2, acc3};
#pragma unroll
    for (int t = 0; t < 4; ++t) {
        int col = t * 16 + m;
        float bv = bias[col];
#pragma unroll
        for (int v = 0; v < 8; ++v) {
            int r = rowBase + v;
            float val = accs[t][v] + bv;
            if (alpha) val = (val >= 0.f) ? val : al * val;
            if (res)   val += res[(size_t)r * CC + col];
            if (Y)  Y[(size_t)r * CC + col] = val;
            if (Yh) Yh[(size_t)r * CC + col] = (__bf16)val;
        }
    }
}

// ===================== host-side orchestration =====================

static char* ws_alloc(char*& p, size_t bytes) {
    char* r = p;
    p += (bytes + 255) & ~(size_t)255;
    return r;
}

static void launch_gcn_layer(const __bf16* inh, int ldin,
                             __bf16* outh, int ldo, int colOff,
                             const __bf16* W, const float* bias,
                             float* hw, const float* snorm,
                             const int* nidx, const float* nw, const int* cnt,
                             hipStream_t s) {
    k_gemm2<<<GEMM_BLOCKS, 128, 0, s>>>(inh, ldin, W, hw, nullptr,
                                        nullptr, nullptr, 64);
    k_gcn_agg<<<cdiv(BNROWS * 64, 256), 256, 0, s>>>(hw, snorm, bias, nidx, nw, cnt,
                                                     outh, ldo, colOff);
}

extern "C" void kernel_launch(void* const* d_in, const int* in_sizes, int n_in,
                              void* d_out, int out_size, void* d_ws, size_t ws_size,
                              hipStream_t stream) {
    const float* x      = (const float*)d_in[0];
    const float* gcn_w  = (const float*)d_in[1];
    const float* gcn_b  = (const float*)d_in[2];
    const float* cw     = (const float*)d_in[3];
    const float* cb     = (const float*)d_in[4];
    const float* rb_w   = (const float*)d_in[5];
    const float* rb_b   = (const float*)d_in[6];
    const float* prelu  = (const float*)d_in[7];
    const int*   eidx   = (const int*)d_in[8];
    int E = in_sizes[8] / 2;
    const int* src = eidx;
    const int* dst = eidx + E;

    // ---- workspace carve ----
    char* p = (char*)d_ws;
    float*  A     = (float*)ws_alloc(p, (size_t)BNROWS * 64 * 4);     // f32 activation ping
    float*  T2    = (float*)ws_alloc(p, (size_t)BNROWS * 64 * 4);     // f32 activation pong
    float*  T1    = (float*)ws_alloc(p, (size_t)BNROWS * 64 * 4);     // f32 hw (GCN gemm out)
    __bf16* Ah    = (__bf16*)ws_alloc(p, (size_t)BNROWS * 64 * 2);    // bf16 mirror ping
    __bf16* T2h   = (__bf16*)ws_alloc(p, (size_t)BNROWS * 64 * 2);    // bf16 mirror pong
    __bf16* T1h   = (__bf16*)ws_alloc(p, (size_t)BNROWS * 64 * 2);    // bf16 conv intermediate
    __bf16* Fh    = (__bf16*)ws_alloc(p, (size_t)BNROWS * 192 * 2);   // bf16 concat buffer
    float*  deg   = (float*)ws_alloc(p, (size_t)NP * 4);              // deg -> dis
    float*  snorm = (float*)ws_alloc(p, (size_t)NP * 4);
    int*    cnt   = (int*)  ws_alloc(p, (size_t)NP * 4);
    int*    nidx  = (int*)  ws_alloc(p, (size_t)NP * 8 * 4);
    float*  nw    = (float*)ws_alloc(p, (size_t)NP * 8 * 4);
    __bf16* gW    = (__bf16*)ws_alloc(p, (size_t)27 * 64 * 64 * 2);
    __bf16* cWt   = (__bf16*)ws_alloc(p, (size_t)3 * 64 * 192 * 2);
    __bf16* rWt   = (__bf16*)ws_alloc(p, (size_t)16 * 9 * 64 * 64 * 2);

    // ---- per-launch setup ----
    k_cvt_gcn_w<<<cdiv(27 * 64 * 64, 256), 256, 0, stream>>>(gcn_w, gW);
    k_cvt_cw<<<cdiv(3 * 64 * 192, 256), 256, 0, stream>>>(cw, cWt);
    k_cvt_rb<<<cdiv(16 * 9 * 64 * 64, 256), 256, 0, stream>>>(rb_w, rWt);
    k_deg_init<<<cdiv(NP, 256), 256, 0, stream>>>(deg);
    k_deg_count<<<cdiv(E, 256), 256, 0, stream>>>(deg, dst, E);
    k_norm_fin<<<cdiv(NP, 256), 256, 0, stream>>>(deg, snorm);
    k_zero_cnt<<<cdiv(NP, 256), 256, 0, stream>>>(cnt);
    k_adj_build<<<cdiv(E, 256), 256, 0, stream>>>(src, dst, deg, cnt, nidx, nw, E);
    k_nchw2rows<<<cdiv(BB * CC * NP, 256), 256, 0, stream>>>(x, A, Ah);

    float*  cur = A;     __bf16* curh = Ah;
    float*  spare = T2;  __bf16* spareh = T2h;

    for (int stg = 0; stg < 3; ++stg) {
        // --- 3 DAM branches -> Fh[:, j*64 : (j+1)*64] ---
        for (int j = 0; j < 3; ++j) {
            int inst = 3 * stg + j;
            const __bf16* W0 = gW + (size_t)(inst * 3 + 0) * 64 * 64;
            const __bf16* W1 = gW + (size_t)(inst * 3 + 1) * 64 * 64;
            const __bf16* W2 = gW + (size_t)(inst * 3 + 2) * 64 * 64;
            const float* b0 = gcn_b + (size_t)(inst * 3 + 0) * 64;
            const float* b1 = gcn_b + (size_t)(inst * 3 + 1) * 64;
            const float* b2 = gcn_b + (size_t)(inst * 3 + 2) * 64;
            // L0: curh -> Fh col j
            launch_gcn_layer(curh, 64, Fh, 192, j * 64, W0, b0, T1, snorm, nidx, nw, cnt, stream);
            // L1: Fh col j -> spareh
            launch_gcn_layer(Fh + j * 64, 192, spareh, 64, 0, W1, b1, T1, snorm, nidx, nw, cnt, stream);
            // L2: spareh -> Fh col j
            launch_gcn_layer(spareh, 64, Fh, 192, j * 64, W2, b2, T1, snorm, nidx, nw, cnt, stream);
        }
        // --- 1x1 fusion conv (K=192) + bias + residual(cur) -> spare(+mirror) ---
        k_gemm2<<<GEMM_BLOCKS, 128, 0, stream>>>(Fh, 192, cWt + (size_t)stg * 64 * 192,
                                                 spare, spareh, cb + stg * 64, cur, 192);
        { float* t = cur; cur = spare; spare = t; }
        { __bf16* t = curh; curh = spareh; spareh = t; }

        // --- ResBlock group after stages 0 and 1 ---
        if (stg < 2) {
            for (int blk = 0; blk < 4; ++blk) {
                int conv0 = (stg * 4 + blk) * 2;
                // conv1 + bias + PReLU -> T1h (bf16 only; consumed only as GEMM input)
                k_conv2<<<GEMM_BLOCKS, 128, 0, stream>>>(curh, rWt + (size_t)conv0 * 9 * 64 * 64,
                                                         nullptr, T1h,
                                                         rb_b + (size_t)conv0 * 64,
                                                         prelu + (stg * 4 + blk), nullptr);
                // conv2 + bias + residual(cur) -> spare(+mirror)
                k_conv2<<<GEMM_BLOCKS, 128, 0, stream>>>(T1h, rWt + (size_t)(conv0 + 1) * 9 * 64 * 64,
                                                         spare, spareh,
                                                         rb_b + (size_t)(conv0 + 1) * 64,
                                                         nullptr, cur);
                { float* t = cur; cur = spare; spare = t; }
                { __bf16* t = curh; curh = spareh; spareh = t; }
            }
        }
    }

    k_rows2nchw<<<cdiv(BB * CC * NP, 256), 256, 0, stream>>>(cur, (float*)d_out);
}
